// WhisperMQAttention_51247549776245
// MI455X (gfx1250) — matrix operand order repaired and checked by reading, not compiled, not run on hardware
//
#include <hip/hip_runtime.h>
#include <hip/hip_bf16.h>

// ---------------------------------------------------------------------------
// WhisperMQAttention for MI455X (gfx1250, wave32, WMMA).
// f32->bf16 cast; Q/KV/O projections via v_wmma_f32_16x16x32_bf16 with 64x64
// tiles per wave; flash attention with 64-key chunks where the softmax row
// reduction runs in the A-fragment domain (S transposed through LDS): in-lane
// VALU reductions + a single shfl_xor(16), and P is packed into WMMA A
// operands directly in registers.
// ---------------------------------------------------------------------------

typedef __bf16 bf16;
typedef __attribute__((ext_vector_type(16))) __bf16 v16bf;
typedef __attribute__((ext_vector_type(8)))  __bf16 v8bf;
typedef __attribute__((ext_vector_type(8)))  float  v8f;
typedef __attribute__((ext_vector_type(4)))  float  v4f;

#define EMBED   1024
#define T_LEN   2048
#define BATCH   2
#define HDIM    64
#define NHEADS  16
#define MROWS   (BATCH * T_LEN)      // 4096 tokens total
#define KVDIM   (2 * HDIM)           // 128
#define LOG2E   1.44269504088896340736f

static __device__ __forceinline__ v16bf cat8(v8bf lo, v8bf hi) {
  v16bf r;
#pragma unroll
  for (int i = 0; i < 8; ++i) { r[i] = lo[i]; r[i + 8] = hi[i]; }
  return r;
}

static __device__ __forceinline__ v8f wmma_bf16(v16bf a, v16bf b, v8f c) {
  // D(16x16,f32) = A(16x32,bf16) * B(32x16,bf16) + C
  return __builtin_amdgcn_wmma_f32_16x16x32_bf16(
      /*neg_a=*/false, a, /*neg_b=*/false, b,
      /*c_mod=*/(short)0, c, /*reuse_a=*/false, /*reuse_b=*/false);
}

// A-fragment (16x32): row-major src, row = lane&15; lane<16 holds K{0-7,16-23},
// lane>=16 holds K{8-15,24-31} (CDNA5 ISA 16-bit A layout).
static __device__ __forceinline__ v16bf load_a(const bf16* A, int rowBase,
                                               int lda, int kk, int lane) {
  int row = lane & 15, kh = lane >> 4;
  const v8bf* p = (const v8bf*)(A + (size_t)(rowBase + row) * lda + kk);
  return cat8(p[kh], p[2 + kh]);
}

// B-fragment (32x16) with B[k][n] = W[n][k], W row-major (N x K):
// column n = lane&15; lanes<16 hold K{0-7,16-23}, lanes>=16 hold K{8-15,24-31}.
static __device__ __forceinline__ v16bf load_b(const bf16* W, int colBase,
                                               int ldk, int kk, int lane) {
  int col = lane & 15, kg = lane >> 4;
  const v8bf* p = (const v8bf*)(W + (size_t)(colBase + col) * ldk + kk + 8 * kg);
  return cat8(p[0], p[2]);
}

// ---------------------------------------------------------------------------
__global__ void f32_to_bf16(const float* __restrict__ in, bf16* __restrict__ out,
                            int n) {
  int i = blockIdx.x * blockDim.x + threadIdx.x;
  if (i < n) out[i] = (bf16)in[i];
}

// ---------------------------------------------------------------------------
// C(MxN) = A(MxK) * W(NxK)^T + bias, 64x64 tile per wave (16 accumulators).
// MODE 0: fp32 out.  MODE 1: bf16 out, *scale.  MODE 2: bf16 kv out + V^T side
// store (vT[b][d][s], d = n-64) straight from the accumulator C-layout.
template <int MODE>
__global__ void __launch_bounds__(32, 1)
gemm_xwT(const bf16* __restrict__ A, const bf16* __restrict__ W,
         const float* __restrict__ bias, void* __restrict__ out,
         bf16* __restrict__ vT, int M, int N, int K, float scale) {
  int lane = threadIdx.x & 31;
  int n0 = blockIdx.x * 64, m0 = blockIdx.y * 64;

  v8f acc[4][4] = {};
  for (int kk = 0; kk < K; kk += 32) {
    // prefetch next A chunk (global_prefetch_b8; speculative, OOB is dropped)
    __builtin_prefetch(A + (size_t)(m0 + (lane & 15)) * K + kk + 64, 0, 1);
    v16bf a[4], b[4];
#pragma unroll
    for (int i = 0; i < 4; ++i) a[i] = load_a(A, m0 + 16 * i, K, kk, lane);
#pragma unroll
    for (int j = 0; j < 4; ++j) b[j] = load_b(W, n0 + 16 * j, K, kk, lane);
#pragma unroll
    for (int i = 0; i < 4; ++i)
#pragma unroll
      for (int j = 0; j < 4; ++j)
        acc[i][j] = wmma_bf16(a[i], b[j], acc[i][j]);
  }

  // C-layout: VGPR g, lane l -> M = g + 8*(l>>4), N = l&15
  int col = lane & 15, rg = lane >> 4;
#pragma unroll
  for (int nj = 0; nj < 4; ++nj) {
    int n = n0 + nj * 16 + col;
    float bv = bias[n];
#pragma unroll
    for (int mi = 0; mi < 4; ++mi) {
#pragma unroll
      for (int g = 0; g < 8; ++g) {
        int m = m0 + mi * 16 + g + 8 * rg;
        float v = acc[mi][nj][g] + bv;
        if (MODE == 0) {
          ((float*)out)[(size_t)m * N + n] = v;
        } else {
          bf16 bvv = (bf16)(v * scale);
          ((bf16*)out)[(size_t)m * N + n] = bvv;
          if (MODE == 2 && n >= HDIM) {   // V^T[b][d][s] for the PV WMMA path
            int bb = m / T_LEN, s = m % T_LEN;
            vT[((size_t)(bb * HDIM + (n - HDIM))) * T_LEN + s] = bvv;
          }
        }
      }
    }
  }
}

// ---------------------------------------------------------------------------
// Flash MQA: one wave per 16-query-row tile; 64-key chunks. Because
// qr[b,r,:] = q[b, r/16, (r%16)*64 : ...], a 16-row tile with r0 = 16*tq is
// exactly token tq across all 16 heads -> Q fragment is one contiguous row.
// q was pre-scaled by 1/sqrt(HDIM) * log2(e) so softmax uses exp2 natively.
// Output remap replicates the reference reshape: r -> (h = r/T, t = r%T).
__global__ void __launch_bounds__(32, 1)
mqa_flash(const bf16* __restrict__ q, const bf16* __restrict__ kv,
          const bf16* __restrict__ vT, bf16* __restrict__ attn) {
  int lane = threadIdx.x & 31;
  int b = blockIdx.y;
  int r0 = blockIdx.x * 16;   // query-row tile base, r0 in [0, 16*T)
  int tq = r0 >> 4;           // shared token for this tile

  __shared__ __align__(16) float ST[16][68];  // S transpose staging (padded)
  __shared__ __align__(16) float SCF[16];     // per-row rescale broadcast
  __shared__ __align__(16) float LF[16];      // per-row 1/L broadcast

  int col = lane & 15, kg = lane >> 4;
  int row = col;              // A-domain row owned by this lane (dup in halves)

  // Q A-fragments (rows = heads 0..15 of token tq), d chunks {0..31},{32..63}
  const bf16* qrow = q + ((size_t)(b * T_LEN + tq)) * EMBED + col * HDIM;
  const v8bf* qp = (const v8bf*)qrow;
  v16bf qa0 = cat8(qp[kg],     qp[2 + kg]);
  v16bf qa1 = cat8(qp[4 + kg], qp[6 + kg]);

  float Mrun = -1e30f, Lrun = 0.f;   // per-lane row stats (log2 domain)
  v8f o[4] = {};

  for (int s0 = 0; s0 < T_LEN; s0 += 64) {
    // ---- S = Q @ K^T for 64 keys (four 16-key N-tiles) ----
    v8f sacc[4] = {};
#pragma unroll
    for (int j = 0; j < 4; ++j) {
      const v8bf* kp =
          (const v8bf*)(kv + ((size_t)(b * T_LEN + s0 + j * 16 + col)) * KVDIM);
      v16bf kb0 = cat8(kp[kg],     kp[2 + kg]);     // d = 0..31
      v16bf kb1 = cat8(kp[4 + kg], kp[6 + kg]);     // d = 32..63
      sacc[j] = wmma_bf16(qa0, kb0, sacc[j]);
      sacc[j] = wmma_bf16(qa1, kb1, sacc[j]);
    }

    // ---- transpose S to LDS (C-layout -> row-major) ----
#pragma unroll
    for (int j = 0; j < 4; ++j)
#pragma unroll
      for (int g = 0; g < 8; ++g)
        ST[g + 8 * kg][j * 16 + col] = sacc[j][g];
    __syncthreads();

    // ---- A-domain softmax: this lane owns 32 of row's 64 scores ----
    const float* srow = &ST[row][0];
    float sv[32];
#pragma unroll
    for (int grp = 0; grp < 4; ++grp) {
      const v4f* gp = (const v4f*)(srow + 16 * grp + 8 * kg);
      v4f x0 = gp[0], x1 = gp[1];
#pragma unroll
      for (int e = 0; e < 4; ++e) { sv[grp * 8 + e] = x0[e]; sv[grp * 8 + 4 + e] = x1[e]; }
    }
    float mx = sv[0];
#pragma unroll
    for (int i = 1; i < 32; ++i) mx = fmaxf(mx, sv[i]);
    mx = fmaxf(mx, __shfl_xor(mx, 16, 32));     // combine the two half-rows
    float mn = fmaxf(Mrun, mx);
    float sc = exp2f(Mrun - mn);
    Mrun = mn;

    // exp2 + pack P directly into WMMA A-operands (register-resident)
    v16bf pa0, pa1;
    float rs = 0.f;
#pragma unroll
    for (int i = 0; i < 16; ++i) {
      float p = exp2f(sv[i] - mn);
      rs += p; pa0[i] = (bf16)p;
    }
#pragma unroll
    for (int i = 0; i < 16; ++i) {
      float p = exp2f(sv[16 + i] - mn);
      rs += p; pa1[i] = (bf16)p;
    }
    rs += __shfl_xor(rs, 16, 32);
    Lrun = Lrun * sc + rs;

    // ---- broadcast sc to C-layout rows, rescale O ----
    SCF[row] = sc;                               // both halves write same value
    __syncthreads();
    const v4f* scp = (const v4f*)&SCF[8 * kg];
    v4f sl0 = scp[0], sl1 = scp[1];
#pragma unroll
    for (int dt = 0; dt < 4; ++dt)
#pragma unroll
      for (int g = 0; g < 8; ++g)
        o[dt][g] *= (g < 4) ? sl0[g & 3] : sl1[g & 3];

    // ---- O += P @ V (V^T gives contiguous B-fragments) ----
#pragma unroll
    for (int dt = 0; dt < 4; ++dt) {
      const bf16* vcol = vT + ((size_t)(b * HDIM + dt * 16 + col)) * T_LEN;
      const v8bf* vp0 = (const v8bf*)(vcol + s0 + 8 * kg);
      const v8bf* vp1 = (const v8bf*)(vcol + s0 + 32 + 8 * kg);
      o[dt] = wmma_bf16(pa0, cat8(vp0[0], vp0[2]), o[dt]);
      o[dt] = wmma_bf16(pa1, cat8(vp1[0], vp1[2]), o[dt]);
    }
  }

  // ---- normalize (broadcast 1/L) + store with head/token remap ----
  LF[row] = 1.f / Lrun;
  __syncthreads();
  const v4f* lp = (const v4f*)&LF[8 * kg];
  v4f ll0 = lp[0], ll1 = lp[1];

  int h = r0 / T_LEN;          // constant over the 16-row tile
  int tbase = r0 % T_LEN;
#pragma unroll
  for (int dt = 0; dt < 4; ++dt)
#pragma unroll
    for (int g = 0; g < 8; ++g) {
      int t = tbase + g + 8 * kg;
      float inv = (g < 4) ? ll0[g & 3] : ll1[g & 3];
      attn[((size_t)(b * T_LEN + t)) * EMBED + h * HDIM + dt * 16 + col] =
          (bf16)(o[dt][g] * inv);
    }
}

// ---------------------------------------------------------------------------
extern "C" void kernel_launch(void* const* d_in, const int* in_sizes, int n_in,
                              void* d_out, int out_size, void* d_ws,
                              size_t ws_size, hipStream_t stream) {
  (void)in_sizes; (void)n_in; (void)out_size; (void)ws_size;
  const float* x   = (const float*)d_in[0];
  const float* Wq  = (const float*)d_in[1];
  const float* bq  = (const float*)d_in[2];
  const float* Wkv = (const float*)d_in[3];
  const float* bkv = (const float*)d_in[4];
  const float* Wo  = (const float*)d_in[5];
  const float* bo  = (const float*)d_in[6];
  float* out = (float*)d_out;

  // Workspace layout (bf16 elements), ~30 MB total
  bf16* w = (bf16*)d_ws;
  size_t off = 0;
  bf16* x_bf   = w + off; off += (size_t)MROWS * EMBED;
  bf16* wq_bf  = w + off; off += (size_t)EMBED * EMBED;
  bf16* wkv_bf = w + off; off += (size_t)KVDIM * EMBED;
  bf16* wo_bf  = w + off; off += (size_t)EMBED * EMBED;
  bf16* q_bf   = w + off; off += (size_t)MROWS * EMBED;
  bf16* kv_bf  = w + off; off += (size_t)MROWS * KVDIM;
  bf16* vT_bf  = w + off; off += (size_t)BATCH * HDIM * T_LEN;
  bf16* at_bf  = w + off; off += (size_t)MROWS * EMBED;

  auto conv = [&](const float* src, bf16* dst, int n) {
    f32_to_bf16<<<(n + 255) / 256, 256, 0, stream>>>(src, dst, n);
  };
  conv(x,   x_bf,   MROWS * EMBED);
  conv(Wq,  wq_bf,  EMBED * EMBED);
  conv(Wkv, wkv_bf, KVDIM * EMBED);
  conv(Wo,  wo_bf,  EMBED * EMBED);

  // q = (x @ Wq^T + bq) * 1/sqrt(HDIM) * log2(e)  (exp2-domain softmax)
  gemm_xwT<1><<<dim3(EMBED / 64, MROWS / 64), 32, 0, stream>>>(
      x_bf, wq_bf, bq, q_bf, nullptr, MROWS, EMBED, EMBED, 0.125f * LOG2E);
  // kv = x @ Wkv^T + bkv  (+ V^T side store)
  gemm_xwT<2><<<dim3(KVDIM / 64, MROWS / 64), 32, 0, stream>>>(
      x_bf, wkv_bf, bkv, kv_bf, vT_bf, MROWS, KVDIM, EMBED, 1.0f);
  // flash MQA over all (batch, 16-row query tiles)
  mqa_flash<<<dim3((NHEADS * T_LEN) / 16, BATCH), 32, 0, stream>>>(
      q_bf, kv_bf, vT_bf, at_bf);
  // out = attn @ Wo^T + bo  (fp32)
  gemm_xwT<0><<<dim3(EMBED / 64, MROWS / 64), 32, 0, stream>>>(
      at_bf, wo_bf, bo, out, nullptr, MROWS, EMBED, EMBED, 1.0f);
}
